// TransformerBlock_43112881717669
// MI455X (gfx1250) — matrix operand order repaired and checked by reading, not compiled, not run on hardware
//
#include <hip/hip_runtime.h>
#include <hip/hip_bf16.h>

#define D_MODEL 1024
#define N_HEADS 16
#define D_K     64
#define D_FF    4096
#define SEQ     2048

#define LDSB_STRIDE 40   // 32 halfs + 8 pad: 80B row stride -> conflict-free ds_load_b128

typedef _Float16 half8  __attribute__((ext_vector_type(8)));
typedef _Float16 half16 __attribute__((ext_vector_type(16)));
typedef float    float8 __attribute__((ext_vector_type(8)));
typedef int      int4v  __attribute__((ext_vector_type(4)));

#define AS1 __attribute__((address_space(1)))
#define AS3 __attribute__((address_space(3)))

// Async global->LDS path (CDNA5): guarded so we fall back cleanly if the
// toolchain lacks the builtin.
#if defined(__gfx1250__) && __has_builtin(__builtin_amdgcn_global_load_async_to_lds_b128)
#define GEMM_ASYNC 1
#else
#define GEMM_ASYNC 0
#endif

#if GEMM_ASYNC
#if __has_builtin(__builtin_amdgcn_s_wait_asynccnt)
#define WAIT_ASYNC(n) __builtin_amdgcn_s_wait_asynccnt(n)
#else
#define WAIT_ASYNC(n) asm volatile("s_wait_asynccnt " #n ::: "memory")
#endif
#else
#define WAIT_ASYNC(n)
#endif

// ---------------------------------------------------------------------------
// WMMA helpers (gfx1250, wave32). Layouts used here (A/B k order is this code's own choice, not the ISA table's):
//  A (16x32 f16): lane L -> row L%16; halfs 0..15 = K (L/16)*16 .. +15 (contiguous)
//  B (32x16 f16): lane L -> col L%16; halfs 0..15 = K (L/16)*16 .. +15 (contiguous)
//  C/D (16x16 f32): lane L -> col L%16; VGPR i -> row i + 8*(L/16)
// ---------------------------------------------------------------------------
__device__ __forceinline__ float8 wmma_f16(half16 a, half16 b, float8 c) {
  return __builtin_amdgcn_wmma_f32_16x16x32_f16(false, a, false, b, (short)0, c,
                                                false, false);
}

__device__ __forceinline__ half16 frag_a(const _Float16* base, int row0, int ld,
                                         int kb, int lane) {
  int r  = row0 + (lane & 15);
  int ks = (lane >> 4) * 16;
  const _Float16* p = base + (size_t)r * ld + kb + ks;
  half8 lo = *(const half8*)p;
  half8 hi = *(const half8*)(p + 8);
  return __builtin_shufflevector(lo, hi, 0, 1, 2, 3, 4, 5, 6, 7,
                                 8, 9, 10, 11, 12, 13, 14, 15);
}

__device__ __forceinline__ half16 frag_b(const _Float16* base, int col0, int ld,
                                         int kb, int lane) {
  int r = col0 + (lane & 15);
  int k = kb + (lane >> 4) * 16;
  const _Float16* p = base + (size_t)r * ld + k;
  half8 lo = *(const half8*)p;
  half8 hi = *(const half8*)(p + 8);
  return __builtin_shufflevector(lo, hi, 0, 1, 2, 3, 4, 5, 6, 7,
                                 8, 9, 10, 11, 12, 13, 14, 15);
}

__device__ __forceinline__ half16 frag_b_lds(const _Float16* ldsb, int col0, int lane) {
  int r = col0 + (lane & 15);
  int k = (lane >> 4) * 16;
  const _Float16* p = ldsb + r * LDSB_STRIDE + k;
  half8 lo = *(const half8*)p;
  half8 hi = *(const half8*)(p + 8);
  return __builtin_shufflevector(lo, hi, 0, 1, 2, 3, 4, 5, 6, 7,
                                 8, 9, 10, 11, 12, 13, 14, 15);
}

// Stage a 64(col) x 32(K) f16 B-tile into LDS: 256 x 16B chunks, 128 threads.
__device__ __forceinline__ void stage_b(const _Float16* __restrict__ Bt, int n0, int K,
                                        int kb, int tid, _Float16* dst) {
#pragma unroll
  for (int j = 0; j < 2; ++j) {
    int c = tid * 2 + j;   // chunk 0..255
    int r = c >> 2;        // tile row 0..63
    int part = c & 3;      // 16B chunk within 64B row
    const _Float16* g = Bt + (size_t)(n0 + r) * K + kb + part * 8;
    _Float16* l = dst + r * LDSB_STRIDE + part * 8;
#if GEMM_ASYNC
    AS1 const void* gv = (AS1 const void*)g;  // addrspacecast generic -> global
    AS3 void*       lv = (AS3 void*)l;        // addrspacecast generic -> lds
    __builtin_amdgcn_global_load_async_to_lds_b128((AS1 int4v*)gv, (AS3 int4v*)lv, 0, 0);
#else
    *(half8*)l = *(const half8*)g;
#endif
  }
}

// ---------------------------------------------------------------------------
// LayerNorm (f32 in) -> f16 out. One block per row of D_MODEL.
// ---------------------------------------------------------------------------
__global__ void ln_to_f16(const float* __restrict__ x, const float* __restrict__ g,
                          const float* __restrict__ be, _Float16* __restrict__ out) {
  __shared__ float red[256];
  int row = blockIdx.x;
  const float* xr = x + (size_t)row * D_MODEL;
  float s = 0.f;
  for (int i = threadIdx.x; i < D_MODEL; i += 256) s += xr[i];
  red[threadIdx.x] = s;
  __syncthreads();
  for (int off = 128; off > 0; off >>= 1) {
    if (threadIdx.x < off) red[threadIdx.x] += red[threadIdx.x + off];
    __syncthreads();
  }
  float mu = red[0] * (1.0f / D_MODEL);
  __syncthreads();
  float v = 0.f;
  for (int i = threadIdx.x; i < D_MODEL; i += 256) {
    float d = xr[i] - mu;
    v += d * d;
  }
  red[threadIdx.x] = v;
  __syncthreads();
  for (int off = 128; off > 0; off >>= 1) {
    if (threadIdx.x < off) red[threadIdx.x] += red[threadIdx.x + off];
    __syncthreads();
  }
  float rstd = rsqrtf(red[0] * (1.0f / D_MODEL) + 1e-5f);
  _Float16* orow = out + (size_t)row * D_MODEL;
  for (int i = threadIdx.x; i < D_MODEL; i += 256)
    orow[i] = (_Float16)((xr[i] - mu) * rstd * g[i] + be[i]);
}

// ---------------------------------------------------------------------------
// Weight prep: W (K x N, f32) -> Wt (N x K, f16), LDS-tiled coalesced transpose.
// Grid (K/32, N/32), block (32, 8).
// ---------------------------------------------------------------------------
__global__ void w_transpose_f16(const float* __restrict__ W, _Float16* __restrict__ Wt,
                                int K, int N) {
  __shared__ float tile[32][33];
  int k0 = blockIdx.x * 32;
  int n0 = blockIdx.y * 32;
  int tx = threadIdx.x, ty = threadIdx.y;
#pragma unroll
  for (int j = 0; j < 32; j += 8)
    tile[ty + j][tx] = W[(size_t)(k0 + ty + j) * N + (n0 + tx)];
  __syncthreads();
#pragma unroll
  for (int j = 0; j < 32; j += 8)
    Wt[(size_t)(n0 + ty + j) * K + (k0 + tx)] = (_Float16)tile[tx][ty + j];
}

// ---------------------------------------------------------------------------
// WMMA GEMM: C(M x N) = A(M x K, f16) @ Bt(N x K, f16)^T + bias
// Block: 128 threads = 4 waves; each wave -> 32 rows x 64 cols (8 wmma / K-step).
// B tile (64 x 32) staged in LDS with double-buffered async copies.
// Grid: (N/64, M/128). M % 128 == 0, N % 64 == 0 for all call sites.
// ---------------------------------------------------------------------------
#define MODE_F16    0  // outh row-major f16 (optionally GELU)
#define MODE_BHSD   1  // outh as (B, H, S, Dk) f16  (Q / K)
#define MODE_BHDS   2  // outh as (B, H, Dk, S) f16  (V transposed)
#define MODE_RESF32 3  // outf = res + val (f32)

__global__ void __launch_bounds__(128)
gemm_wmma(const _Float16* __restrict__ A, const _Float16* __restrict__ Bt,
          const float* __restrict__ bias, const float* __restrict__ res,
          _Float16* __restrict__ outh, float* __restrict__ outf,
          int M, int N, int K, int mode, int act_gelu) {
  __shared__ __align__(16) _Float16 ldsB[2][64 * LDSB_STRIDE];
  int tid  = threadIdx.x;
  int lane = tid & 31;
  int wave = tid >> 5;
  int m0 = blockIdx.y * 128 + wave * 32;
  int n0 = blockIdx.x * 64;

  float8 acc0[4] = {}, acc1[4] = {};

  stage_b(Bt, n0, K, 0, tid, &ldsB[0][0]);
  int stg = 0;
  for (int kb = 0; kb < K; kb += 32, stg ^= 1) {
    bool more = (kb + 32) < K;
    if (more) {
      stage_b(Bt, n0, K, kb + 32, tid, &ldsB[stg ^ 1][0]);
      // prefetch next A rows for this wave
      __builtin_prefetch(A + (size_t)(m0 + (lane & 15)) * K + kb + 32, 0, 3);
      __builtin_prefetch(A + (size_t)(m0 + 16 + (lane & 15)) * K + kb + 32, 0, 3);
      WAIT_ASYNC(2);   // current stage's 2 async ops done; next stage in flight
    } else {
      WAIT_ASYNC(0);
    }
    __syncthreads();
    half16 a0 = frag_a(A, m0, K, kb, lane);
    half16 a1 = frag_a(A, m0 + 16, K, kb, lane);
#pragma unroll
    for (int t = 0; t < 4; ++t) {
      half16 b = frag_b_lds(&ldsB[stg][0], t * 16, lane);
      acc0[t] = wmma_f16(a0, b, acc0[t]);
      acc1[t] = wmma_f16(a1, b, acc1[t]);
    }
    __syncthreads();
  }

  int cc = lane & 15;
  int rb = (lane >> 4) * 8;
#pragma unroll
  for (int half = 0; half < 2; ++half) {
#pragma unroll
    for (int t = 0; t < 4; ++t) {
      int c = n0 + t * 16 + cc;
      float bv = bias[c];
#pragma unroll
      for (int i = 0; i < 8; ++i) {
        int m = m0 + half * 16 + rb + i;
        float v = (half ? acc1[t][i] : acc0[t][i]) + bv;
        if (act_gelu) v = 0.5f * v * (1.0f + erff(v * 0.70710678118654752f));
        if (mode == MODE_F16) {
          outh[(size_t)m * N + c] = (_Float16)v;
        } else if (mode == MODE_BHSD) {
          int b = m >> 11, s2 = m & (SEQ - 1);
          int hh = c >> 6, d = c & (D_K - 1);
          outh[((((size_t)b * N_HEADS + hh) * SEQ) + s2) * D_K + d] = (_Float16)v;
        } else if (mode == MODE_BHDS) {
          int b = m >> 11, s2 = m & (SEQ - 1);
          int hh = c >> 6, d = c & (D_K - 1);
          outh[((((size_t)b * N_HEADS + hh) * D_K) + d) * SEQ + s2] = (_Float16)v;
        } else {
          outf[(size_t)m * N + c] = res[(size_t)m * N + c] + v;
        }
      }
    }
  }
}

// ---------------------------------------------------------------------------
// Flash-style attention. One wave (block of 32) per 16-query tile per (b,h).
// Q, K stored (B,H,S,Dk) f16; Vt stored (B,H,Dk,S) f16.
// Grid: (SEQ/16, N_HEADS, B), block 32.
// ---------------------------------------------------------------------------
__global__ void attn_flash(const _Float16* __restrict__ Q, const _Float16* __restrict__ Kf,
                           const _Float16* __restrict__ Vt, _Float16* __restrict__ O) {
  __shared__ __align__(16) _Float16 plds[16 * 32];
  int lane = threadIdx.x & 31;
  int q0 = blockIdx.x * 16;
  int h  = blockIdx.y;
  int b  = blockIdx.z;
  const _Float16* Qb = Q  + (((size_t)b * N_HEADS + h) * SEQ) * D_K;
  const _Float16* Kb = Kf + (((size_t)b * N_HEADS + h) * SEQ) * D_K;
  const _Float16* Vb = Vt + (((size_t)b * N_HEADS + h) * D_K) * SEQ;

  half16 aq0 = frag_a(Qb, q0, D_K, 0, lane);
  half16 aq1 = frag_a(Qb, q0, D_K, 32, lane);

  float8 acc[4] = {};
  float rm[8], rs[8];
#pragma unroll
  for (int i = 0; i < 8; ++i) { rm[i] = -1e30f; rs[i] = 0.f; }

  for (int kt = 0; kt < SEQ; kt += 32) {
    float8 s0 = {}, s1 = {};
    s0 = wmma_f16(aq0, frag_b(Kb, kt, D_K, 0, lane), s0);
    s0 = wmma_f16(aq1, frag_b(Kb, kt, D_K, 32, lane), s0);
    s1 = wmma_f16(aq0, frag_b(Kb, kt + 16, D_K, 0, lane), s1);
    s1 = wmma_f16(aq1, frag_b(Kb, kt + 16, D_K, 32, lane), s1);

#pragma unroll
    for (int i = 0; i < 8; ++i) {
      float a0 = s0[i] * 0.125f;  // 1/sqrt(64)
      float a1 = s1[i] * 0.125f;
      float tm = fmaxf(a0, a1);
      for (int msk = 1; msk < 16; msk <<= 1) tm = fmaxf(tm, __shfl_xor(tm, msk, 32));
      float nm = fmaxf(rm[i], tm);
      float corr = __expf(rm[i] - nm);
      float p0 = __expf(a0 - nm);
      float p1 = __expf(a1 - nm);
      float ps = p0 + p1;
      for (int msk = 1; msk < 16; msk <<= 1) ps += __shfl_xor(ps, msk, 32);
      rs[i] = rs[i] * corr + ps;
      rm[i] = nm;
      acc[0][i] *= corr; acc[1][i] *= corr; acc[2][i] *= corr; acc[3][i] *= corr;
      int row = (lane >> 4) * 8 + i;
      plds[row * 32 + (lane & 15)]      = (_Float16)p0;
      plds[row * 32 + 16 + (lane & 15)] = (_Float16)p1;
    }
    __syncthreads();
    half16 pf = frag_a(plds, 0, 32, 0, lane);
#pragma unroll
    for (int t = 0; t < 4; ++t)
      acc[t] = wmma_f16(pf, frag_b(Vb, t * 16, SEQ, kt, lane), acc[t]);
    __syncthreads();
  }

  int cc = lane & 15;
  int rb = (lane >> 4) * 8;
#pragma unroll
  for (int i = 0; i < 8; ++i) {
    float inv = 1.0f / rs[i];
    int m = q0 + rb + i;
    size_t rowoff = ((size_t)b * SEQ + m) * D_MODEL + (size_t)h * D_K;
#pragma unroll
    for (int t = 0; t < 4; ++t)
      O[rowoff + t * 16 + cc] = (_Float16)(acc[t][i] * inv);
  }
}

// ---------------------------------------------------------------------------
// Host-side orchestration
// ---------------------------------------------------------------------------
extern "C" void kernel_launch(void* const* d_in, const int* in_sizes, int n_in,
                              void* d_out, int out_size, void* d_ws, size_t ws_size,
                              hipStream_t stream) {
  const float* x     = (const float*)d_in[0];
  const float* Wq    = (const float*)d_in[1];
  const float* bq    = (const float*)d_in[2];
  const float* Wk    = (const float*)d_in[3];
  const float* bk    = (const float*)d_in[4];
  const float* Wv    = (const float*)d_in[5];
  const float* bv    = (const float*)d_in[6];
  const float* Wo    = (const float*)d_in[7];
  const float* bo    = (const float*)d_in[8];
  const float* W1    = (const float*)d_in[9];
  const float* b1    = (const float*)d_in[10];
  const float* W2    = (const float*)d_in[11];
  const float* b2    = (const float*)d_in[12];
  const float* ln1_g = (const float*)d_in[13];
  const float* ln1_b = (const float*)d_in[14];
  const float* ln2_g = (const float*)d_in[15];
  const float* ln2_b = (const float*)d_in[16];

  int rows = in_sizes[0] / D_MODEL;  // B * S
  int Bn   = rows / SEQ;

  char* ws = (char*)d_ws;
  size_t off = 0;
  auto alloc = [&](size_t bytes) -> void* {
    void* p = ws + off;
    off += (bytes + 255) & ~(size_t)255;
    return p;
  };

  _Float16* Wqt  = (_Float16*)alloc((size_t)D_MODEL * D_MODEL * 2);
  _Float16* Wkt  = (_Float16*)alloc((size_t)D_MODEL * D_MODEL * 2);
  _Float16* Wvt  = (_Float16*)alloc((size_t)D_MODEL * D_MODEL * 2);
  _Float16* Wot  = (_Float16*)alloc((size_t)D_MODEL * D_MODEL * 2);
  _Float16* W1t  = (_Float16*)alloc((size_t)D_MODEL * D_FF * 2);
  _Float16* W2t  = (_Float16*)alloc((size_t)D_FF * D_MODEL * 2);
  _Float16* nrm1 = (_Float16*)alloc((size_t)rows * D_MODEL * 2);
  _Float16* Qb   = (_Float16*)alloc((size_t)rows * D_MODEL * 2);
  _Float16* Kb   = (_Float16*)alloc((size_t)rows * D_MODEL * 2);
  _Float16* Vtb  = (_Float16*)alloc((size_t)rows * D_MODEL * 2);
  _Float16* attn = (_Float16*)alloc((size_t)rows * D_MODEL * 2);
  float*    x1   = (float*)   alloc((size_t)rows * D_MODEL * 4);
  _Float16* nrm2 = (_Float16*)alloc((size_t)rows * D_MODEL * 2);
  _Float16* hbuf = (_Float16*)alloc((size_t)rows * D_FF * 2);

  // Weight prep (f32 -> f16, transposed to N x K)
  {
    dim3 tb(32, 8);
    w_transpose_f16<<<dim3(D_MODEL / 32, D_MODEL / 32), tb, 0, stream>>>(Wq, Wqt, D_MODEL, D_MODEL);
    w_transpose_f16<<<dim3(D_MODEL / 32, D_MODEL / 32), tb, 0, stream>>>(Wk, Wkt, D_MODEL, D_MODEL);
    w_transpose_f16<<<dim3(D_MODEL / 32, D_MODEL / 32), tb, 0, stream>>>(Wv, Wvt, D_MODEL, D_MODEL);
    w_transpose_f16<<<dim3(D_MODEL / 32, D_MODEL / 32), tb, 0, stream>>>(Wo, Wot, D_MODEL, D_MODEL);
    w_transpose_f16<<<dim3(D_MODEL / 32, D_FF / 32),    tb, 0, stream>>>(W1, W1t, D_MODEL, D_FF);
    w_transpose_f16<<<dim3(D_FF / 32, D_MODEL / 32),    tb, 0, stream>>>(W2, W2t, D_FF, D_MODEL);
  }

  // LN1
  ln_to_f16<<<rows, 256, 0, stream>>>(x, ln1_g, ln1_b, nrm1);

  // QKV projections
  dim3 gq(D_MODEL / 64, rows / 128);
  gemm_wmma<<<gq, 128, 0, stream>>>(nrm1, Wqt, bq, nullptr, Qb, nullptr,
                                    rows, D_MODEL, D_MODEL, MODE_BHSD, 0);
  gemm_wmma<<<gq, 128, 0, stream>>>(nrm1, Wkt, bk, nullptr, Kb, nullptr,
                                    rows, D_MODEL, D_MODEL, MODE_BHSD, 0);
  gemm_wmma<<<gq, 128, 0, stream>>>(nrm1, Wvt, bv, nullptr, Vtb, nullptr,
                                    rows, D_MODEL, D_MODEL, MODE_BHDS, 0);

  // Attention
  dim3 ga(SEQ / 16, N_HEADS, Bn);
  attn_flash<<<ga, 32, 0, stream>>>(Qb, Kb, Vtb, attn);

  // Output projection + residual -> x1 (f32)
  gemm_wmma<<<gq, 128, 0, stream>>>(attn, Wot, bo, x, nullptr, x1,
                                    rows, D_MODEL, D_MODEL, MODE_RESF32, 0);

  // LN2
  ln_to_f16<<<rows, 256, 0, stream>>>(x1, ln2_g, ln2_b, nrm2);

  // FFN1 with exact GELU -> hbuf (f16)
  dim3 gf1(D_FF / 64, rows / 128);
  gemm_wmma<<<gf1, 128, 0, stream>>>(nrm2, W1t, b1, nullptr, hbuf, nullptr,
                                     rows, D_FF, D_MODEL, MODE_F16, 1);

  // FFN2 + residual -> d_out (f32)
  gemm_wmma<<<gq, 128, 0, stream>>>(hbuf, W2t, b2, x1, nullptr, (float*)d_out,
                                    rows, D_MODEL, D_FF, MODE_RESF32, 0);
}
